// BOW_24781961298234
// MI455X (gfx1250) — matrix operand order repaired and checked by reading, not compiled, not run on hardware
//
#include <hip/hip_runtime.h>

typedef __attribute__((ext_vector_type(16))) _Float16 v16h;
typedef __attribute__((ext_vector_type(8)))  _Float16 v8h;
typedef __attribute__((ext_vector_type(8)))  float    v8f;
typedef __attribute__((ext_vector_type(4)))  float    v4f;

#define B_  32
#define S_  64
#define T_  512
#define D_  128
#define CHUNK 128           // tokens staged in LDS per pass
#define NCHUNK (T_ / CHUNK)
#define GROW  (CHUNK + 16)  // gT row stride (halfs): 288B, multiple of 32B
#define MROW  (T_ + 8)      // Msh row stride (halfs): 1040B, 4B-even, 16B-aligned runs

__global__ __launch_bounds__(256) void bow_wmma_kernel(
    const int*   __restrict__ word_encs,  // [B,T]
    const int*   __restrict__ span_idxs,  // [B,S,2]
    const float* __restrict__ W,          // [V,D]
    const float* __restrict__ bias,       // [D]
    float*       __restrict__ out)        // [B,S,D]
{
    __shared__ __align__(32) _Float16 gT[D_ * GROW];  // W-gather chunk, transposed [d][t_local]
    __shared__ __align__(32) _Float16 Msh[16 * MROW]; // mask matrix [s_local][t], f16 {0,1}
    __shared__ int ids[T_];
    __shared__ int prev[T_];
    __shared__ int sp_i[16], sp_j[16];

    const int b    = blockIdx.x >> 2;   // batch
    const int mt   = blockIdx.x & 3;    // 16-span row tile
    const int tid  = threadIdx.x;
    const int lane = tid & 31;
    const int wave = tid >> 5;          // D-column tile 0..7

    // stage token ids + span bounds
    for (int t = tid; t < T_; t += 256) ids[t] = word_encs[b * T_ + t];
    if (tid < 16) {
        sp_i[tid] = span_idxs[(b * S_ + mt * 16 + tid) * 2 + 0];
        sp_j[tid] = span_idxs[(b * S_ + mt * 16 + tid) * 2 + 1];
    }
    __syncthreads();

    // prev[t] = latest t' < t with same token id, else -1 (dedup: set(1.0) semantics)
    // 4-wide scan: ids[u..u+3] is 16B-aligned -> ds_load_b128
    for (int t = tid; t < T_; t += 256) {
        int id = ids[t], p = -1;
        int u = 0;
        for (; u + 4 <= t; u += 4) {
            if (ids[u + 0] == id) p = u + 0;
            if (ids[u + 1] == id) p = u + 1;
            if (ids[u + 2] == id) p = u + 2;
            if (ids[u + 3] == id) p = u + 3;
        }
        for (; u < t; ++u)
            if (ids[u] == id) p = u;
        prev[t] = p;
    }
    __syncthreads();

    // materialize mask matrix M[s_local][t] once; pack 2 halfs per b32 store
    for (int e = tid; e < 16 * (T_ / 2); e += 256) {
        int sl = e >> 8;           // span-local 0..15
        int t0 = (e & 255) * 2;    // token 0,2,...,510
        int i_s = sp_i[sl], j_s = sp_j[sl];
        int t1 = t0 + 1;
        bool on0 = (i_s <= t0) & (t0 < j_s) & (prev[t0] < i_s);
        bool on1 = (i_s <= t1) & (t1 < j_s) & (prev[t1] < i_s);
        unsigned pk = (on0 ? 0x00003C00u : 0u) | (on1 ? 0x3C000000u : 0u); // {1.0h,1.0h}
        *(unsigned*)(&Msh[sl * MROW + t0]) = pk;
    }
    __syncthreads();

    // per-lane fragment-layout constants (wave32)
    const int m   = lane & 15;               // A-row within tile
    const int akb = (lane < 16) ? 0 : 16;    // A fragment K-base
    const int bkb = (lane < 16) ? 0 : 16;    // B fragment K-base
    const int n   = lane & 15;               // B/C column within tile
    const int d   = wave * 16 + n;           // global output dim

    v8f acc = {};

    for (int chunk = 0; chunk < NCHUNK; ++chunk) {
        const int cb = chunk * CHUNK;

        // prefetch next chunk's gathered W rows into cache (global_prefetch_b8)
        if (chunk + 1 < NCHUNK) {
            const int nb = cb + CHUNK;
            int r0 = tid >> 1, l0 = (tid & 1) * 2;          // rows 0..127, lines {0,1} and {2,3}
            const char* row0 = (const char*)&W[(long long)ids[nb + r0] * D_];
            __builtin_prefetch(row0 + l0 * 128, 0, 1);
            __builtin_prefetch(row0 + (l0 + 1) * 128, 0, 1);
        }

        // gather W rows into LDS as f16, transposed [d][t_local]; float4 coalesced loads
        for (int q = tid; q < CHUNK * D_ / 4; q += 256) {
            int tl = q >> 5;           // token-local 0..127 (32 float4 per 512B row)
            int dq = (q & 31) << 2;    // dim 0,4,...,124
            v4f w4 = *(const v4f*)(&W[(long long)ids[cb + tl] * D_ + dq]);
            _Float16* p = &gT[dq * GROW + tl];
            p[0 * GROW] = (_Float16)w4.x;
            p[1 * GROW] = (_Float16)w4.y;
            p[2 * GROW] = (_Float16)w4.z;
            p[3 * GROW] = (_Float16)w4.w;
        }
        __syncthreads();

        #pragma unroll
        for (int kt = 0; kt < CHUNK / 32; ++kt) {
            // A fragment: two contiguous 8-half runs from the mask matrix
            const int t0 = cb + kt * 32 + akb;
            v8h alo = *(const v8h*)(&Msh[m * MROW + t0]);       // K = akb .. akb+7
            v8h ahi = *(const v8h*)(&Msh[m * MROW + t0 + 8]);   // K = akb+8 .. akb+15
            v16h a = __builtin_shufflevector(alo, ahi,
                        0, 1, 2, 3, 4, 5, 6, 7, 8, 9, 10, 11, 12, 13, 14, 15);

            // B fragment: 16 contiguous halfs from LDS (32B-aligned)
            v16h bf = *(const v16h*)(&gT[d * GROW + kt * 32 + bkb]);

            acc = __builtin_amdgcn_wmma_f32_16x16x32_f16(
                      /*neg_a=*/false, a, /*neg_b=*/false, bf,
                      /*c_mod=*/(short)0, acc,
                      /*reuse_a=*/false, /*reuse_b=*/false);
        }
        __syncthreads();
    }

    // epilogue: C/D layout -> out[b, mt*16 + M, d] + bias
    const float bb = bias[d];
    const int rowbase = (lane < 16) ? 0 : 8;
    #pragma unroll
    for (int r = 0; r < 8; ++r) {
        int srow = mt * 16 + rowbase + r;
        out[((long long)(b * S_ + srow)) * D_ + d] = acc[r] + bb;
    }
}

extern "C" void kernel_launch(void* const* d_in, const int* in_sizes, int n_in,
                              void* d_out, int out_size, void* d_ws, size_t ws_size,
                              hipStream_t stream) {
    (void)in_sizes; (void)n_in; (void)d_ws; (void)ws_size; (void)out_size;
    const int*   word_encs = (const int*)  d_in[0];
    const int*   span_idxs = (const int*)  d_in[1];
    const float* W         = (const float*)d_in[2];
    const float* bias      = (const float*)d_in[3];
    float*       out       = (float*)      d_out;

    dim3 grid(B_ * 4);   // (batch, m-tile)
    dim3 block(256);     // 8 wave32s; wave = D-tile
    bow_wmma_kernel<<<grid, block, 0, stream>>>(word_encs, span_idxs, W, bias, out);
}
